// GraphEnrichedDecoder_82764019794482
// MI455X (gfx1250) — matrix operand order repaired and checked by reading, not compiled, not run on hardware
//
#include <hip/hip_runtime.h>
#include <math.h>

// ---------------------------------------------------------------------------
// GraphEnrichedDecoder for MI455X (gfx1250), wave32.
// Dense GEMMs: v_wmma_f32_16x16x32_f16, 32x32 tile/wave (4 wmma per K-step),
// operands pre-converted to f16, tiles staged to LDS by the Tensor Data Mover
// (tensor_load_to_lds + s_wait_tensorcnt, double buffered).
// up2 fused into convs; final conv+BN+ReLU+1x1 head fused, writes d_out.
// ---------------------------------------------------------------------------

typedef __attribute__((ext_vector_type(16))) _Float16 v16h;
typedef __attribute__((ext_vector_type(8)))  float    v8f;
typedef __attribute__((ext_vector_type(4)))  unsigned v4u;
typedef __attribute__((ext_vector_type(4)))  int      v4i;
typedef __attribute__((ext_vector_type(8)))  int      v8i;

#define N_NODES 256
#define EDGES   4096
#define ETOT    (EDGES + N_NODES)   // self loops appended
#define IN_DIM  1536
#define GH      256
#define HEADS   4
#define HC      64
#define SB      512
#define SS      16

#if defined(__HIP_DEVICE_COMPILE__) && defined(__has_builtin)
#if __has_builtin(__builtin_amdgcn_tensor_load_to_lds) && \
    __has_builtin(__builtin_amdgcn_s_wait_tensorcnt)
#define HAVE_TDM 1
#endif
#endif

// ---------------- utility -------------------------------------------------

__device__ __forceinline__ float wave_sum32(float v) {
#pragma unroll
  for (int o = 16; o > 0; o >>= 1) v += __shfl_xor(v, o, 32);
  return v;
}

__device__ __forceinline__ unsigned f2mono(float f) {
  unsigned u = __float_as_uint(f);
  return (u & 0x80000000u) ? ~u : (u | 0x80000000u);
}
__device__ __forceinline__ float mono2f(unsigned u) {
  return (u & 0x80000000u) ? __uint_as_float(u & 0x7FFFFFFFu)
                           : __uint_as_float(~u);
}

__global__ void zero_f32(float* __restrict__ p, int n) {
  int t = blockIdx.x * blockDim.x + threadIdx.x;
  if (t < n) p[t] = 0.0f;
}

__global__ void cvt_f16(const float* __restrict__ s, _Float16* __restrict__ d,
                        int n) {
  int t = blockIdx.x * blockDim.x + threadIdx.x;
  if (t < n) d[t] = (_Float16)s[t];
}

// ---------------- Tensor Data Mover: 2D f16 tile -> LDS --------------------
// D# per CDNA5 ISA §8: group0 {count=1, lds_addr, global_addr, type=2},
// group1 {data_size=2B, tensor dims == tile dims, dim0 stride}.
#ifdef HAVE_TDM
__device__ __forceinline__ void tdm_load_tile_f16(unsigned lds_addr,
                                                  const _Float16* gptr,
                                                  int stride_elems,
                                                  int tile_w, int tile_h) {
  unsigned long long ga = (unsigned long long)(size_t)gptr;
  v4u g0;
  g0[0] = 1u;                                   // count=1, user mode
  g0[1] = lds_addr;                             // LDS byte address
  g0[2] = (unsigned)(ga & 0xFFFFFFFFu);         // global addr [95:64]
  g0[3] = (unsigned)((ga >> 32) & 0x01FFFFFFu)  // global addr [120:96]
          | (2u << 30);                         // type=2 (image)
  unsigned td0 = (unsigned)tile_w, td1 = (unsigned)tile_h;
  unsigned st0 = (unsigned)stride_elems;
  v8i g1;
  g1[0] = (int)(1u << 16);                      // data_size=1 -> 2 bytes
  g1[1] = (int)((td0 & 0xFFFFu) << 16);         // tensor_dim0 lo16 @[63:48]
  g1[2] = (int)(((td0 >> 16) & 0xFFFFu) |       // tensor_dim0 hi16 @[95:80]->lo
                ((td1 & 0xFFFFu) << 16));       // tensor_dim1 lo16
  g1[3] = (int)(((td1 >> 16) & 0xFFFFu) |       // tensor_dim1 hi16
                ((unsigned)tile_w << 16));      // tile_dim0 @[127:112]
  g1[4] = (int)(td1 & 0xFFFFu);                 // tile_dim1 @[143:128]
  g1[5] = (int)st0;                             // dim0_stride lo32 @[191:160]
  g1[6] = 0;                                    // dim0_stride hi16, dim1_stride lo16
  g1[7] = 0;
  v4i z4 = {0, 0, 0, 0};
#if __clang_major__ >= 23
  v8i z8 = {0, 0, 0, 0, 0, 0, 0, 0};
  __builtin_amdgcn_tensor_load_to_lds(g0, g1, z4, z4, z8, 0);
#else
  __builtin_amdgcn_tensor_load_to_lds(g0, g1, z4, z4, 0);
#endif
}
#endif

// ---------------- WMMA GEMM: C[M,N] = act(A @ B + bias), f16 in, f32 out ---
// One wave computes a 32x32 tile: 4 accumulators, 4 v_wmma per K-step.
// Tiles staged to LDS by TDM (double buffered) or per-lane fallback.
__global__ void __launch_bounds__(32)
wmma_gemm_f16(const _Float16* __restrict__ A, int lda,
              const _Float16* __restrict__ B, int ldb,
              const float* __restrict__ bias,
              float* __restrict__ C, _Float16* __restrict__ Ch, int ldc,
              int K, int act /*0=none,1=gelu(exact)*/) {
  __shared__ __align__(64) _Float16 As[2][32 * 32];  // [m][k]
  __shared__ __align__(64) _Float16 Bs[2][32 * 32];  // [k][n]
  const int m0   = blockIdx.y * 32;
  const int n0   = blockIdx.x * 32;
  const int lane = threadIdx.x;
  const int half = lane >> 4;
  const int mn   = lane & 15;
  const int nk   = K >> 5;

#ifdef HAVE_TDM
#define STAGE(k0, buf)                                                        \
  do {                                                                        \
    tdm_load_tile_f16((unsigned)(size_t)&As[buf][0],                          \
                      A + (size_t)m0 * lda + (k0), lda, 32, 32);              \
    tdm_load_tile_f16((unsigned)(size_t)&Bs[buf][0],                          \
                      B + (size_t)(k0) * ldb + n0, ldb, 32, 32);              \
  } while (0)
#else
#define STAGE(k0, buf)                                                        \
  do {                                                                        \
    _Pragma("unroll")                                                         \
    for (int i_ = 0; i_ < 32; ++i_) {                                         \
      int l_ = lane + 32 * i_;                                                \
      As[buf][l_] = A[(size_t)(m0 + (l_ >> 5)) * lda + (k0) + (l_ & 31)];     \
      Bs[buf][l_] = B[(size_t)((k0) + (l_ >> 5)) * ldb + n0 + (l_ & 31)];     \
    }                                                                         \
  } while (0)
#endif

  v8f acc00 = {}, acc01 = {}, acc10 = {}, acc11 = {};
  STAGE(0, 0);
  for (int i = 0; i < nk; ++i) {
    if (i + 1 < nk) {
      STAGE((i + 1) << 5, (i + 1) & 1);
#ifdef HAVE_TDM
      __builtin_amdgcn_s_wait_tensorcnt(2);  // tile i complete, i+1 in flight
#endif
    } else {
#ifdef HAVE_TDM
      __builtin_amdgcn_s_wait_tensorcnt(0);
#endif
    }
    asm volatile("" ::: "memory");
    const _Float16* as = As[i & 1];
    const _Float16* bs = Bs[i & 1];
    // A frag: half 0 -> K {0..7,16..23}; half 1 -> K {8..15,24..31} (ISA 7.12.2)
    v16h a0, a1, b0, b1;
#pragma unroll
    for (int e = 0; e < 16; ++e) {
      int ka = (e & 7) + ((e >> 3) << 4) + half * 8;
      a0[e] = as[mn * 32 + ka];
      a1[e] = as[(mn + 16) * 32 + ka];
      int kb = (e & 7) + ((e >> 3) << 4) + half * 8;  // B: same K order as A
      b0[e] = bs[kb * 32 + mn];
      b1[e] = bs[kb * 32 + 16 + mn];
    }
    acc00 = __builtin_amdgcn_wmma_f32_16x16x32_f16(false, a0, false, b0,
                                                   (short)0, acc00, false, false);
    acc01 = __builtin_amdgcn_wmma_f32_16x16x32_f16(false, a0, false, b1,
                                                   (short)0, acc01, false, false);
    acc10 = __builtin_amdgcn_wmma_f32_16x16x32_f16(false, a1, false, b0,
                                                   (short)0, acc10, false, false);
    acc11 = __builtin_amdgcn_wmma_f32_16x16x32_f16(false, a1, false, b1,
                                                   (short)0, acc11, false, false);
    asm volatile("" ::: "memory");
  }
#undef STAGE

  // D layout: VGPR i -> row m = 8*half + i, col = lane&15.
  v8f* accs[4] = {&acc00, &acc01, &acc10, &acc11};
#pragma unroll
  for (int s = 0; s < 4; ++s) {
    int mb = m0 + (s >> 1) * 16, nb = n0 + (s & 1) * 16;
    v8f a = *accs[s];
#pragma unroll
    for (int i = 0; i < 8; ++i) {
      int m = mb + half * 8 + i;
      int n = nb + mn;
      float v = a[i] + (bias ? bias[n] : 0.0f);
      if (act == 1) v = 0.5f * v * (1.0f + erff(v * 0.70710678f));
      C[(size_t)m * ldc + n] = v;
      if (Ch) Ch[(size_t)m * ldc + n] = (_Float16)v;
    }
  }
}

// ---------------- LayerNorm over 256 cols (wave per row) -------------------
// out = LN(x + res? + bias?) * gamma + beta; optional f16 shadow copy.
__global__ void __launch_bounds__(32)
ln_row256(const float* __restrict__ x, const float* __restrict__ res,
          const float* __restrict__ bias, const float* __restrict__ gamma,
          const float* __restrict__ beta, float* __restrict__ out,
          _Float16* __restrict__ out_h) {
  const int row  = blockIdx.x;
  const int lane = threadIdx.x;
  float v[8];
  float s = 0.0f;
#pragma unroll
  for (int i = 0; i < 8; ++i) {
    int c = lane + 32 * i;
    float t = x[row * 256 + c];
    if (res)  t += res[row * 256 + c];
    if (bias) t += bias[c];
    v[i] = t;
    s += t;
  }
  float mean = wave_sum32(s) * (1.0f / 256.0f);
  float q = 0.0f;
#pragma unroll
  for (int i = 0; i < 8; ++i) { float d = v[i] - mean; q += d * d; }
  float inv = rsqrtf(wave_sum32(q) * (1.0f / 256.0f) + 1e-5f);
#pragma unroll
  for (int i = 0; i < 8; ++i) {
    int c = lane + 32 * i;
    float o = (v[i] - mean) * inv * gamma[c] + beta[c];
    out[row * 256 + c] = o;
    if (out_h) out_h[row * 256 + c] = (_Float16)o;
  }
}

// ---------------- GATv2 edge kernels --------------------------------------

__device__ __forceinline__ void edge_sd(const int* __restrict__ ei, int e,
                                        int& s, int& d) {
  s = (e < EDGES) ? ei[e]         : (e - EDGES);
  d = (e < EDGES) ? ei[EDGES + e] : (e - EDGES);
}

__global__ void gat_edge_score(const int* __restrict__ ei,
                               const float* __restrict__ xl,
                               const float* __restrict__ xr,
                               const float* __restrict__ att,
                               float* __restrict__ ea,
                               unsigned* __restrict__ em) {
  int t = blockIdx.x * blockDim.x + threadIdx.x;  // e*HEADS + h
  if (t >= ETOT * HEADS) return;
  int e = t >> 2, h = t & 3, s, d;
  edge_sd(ei, e, s, d);
  const float* pl = xl + s * GH + h * HC;
  const float* pr = xr + d * GH + h * HC;
  const float* pa = att + h * HC;
  float a = 0.0f;
#pragma unroll 8
  for (int c = 0; c < HC; ++c) {
    float u = pl[c] + pr[c];
    u = (u > 0.0f) ? u : 0.2f * u;  // leaky_relu 0.2
    a += u * pa[c];
  }
  ea[t] = a;
  atomicMax(&em[d * HEADS + h], f2mono(a));
}

__global__ void gat_edge_exp(const int* __restrict__ ei,
                             const float* __restrict__ ea,
                             const unsigned* __restrict__ em,
                             float* __restrict__ eex,
                             float* __restrict__ es) {
  int t = blockIdx.x * blockDim.x + threadIdx.x;
  if (t >= ETOT * HEADS) return;
  int e = t >> 2, h = t & 3, s, d;
  edge_sd(ei, e, s, d);
  float m = mono2f(em[d * HEADS + h]);
  float x = expf(ea[t] - m);
  eex[t] = x;
  atomicAdd(&es[d * HEADS + h], x);
}

__global__ void gat_scatter(const int* __restrict__ ei,
                            const float* __restrict__ xl,
                            const float* __restrict__ eex,
                            const float* __restrict__ es,
                            float* __restrict__ gout) {
  int t = blockIdx.x * blockDim.x + threadIdx.x;  // e*GH + c
  if (t >= ETOT * GH) return;
  int e = t >> 8, c = t & 255, h = c >> 6, s, d;
  edge_sd(ei, e, s, d);
  float alpha = eex[e * HEADS + h] / (es[d * HEADS + h] + 1e-16f);
  atomicAdd(&gout[d * GH + c], xl[s * GH + c] * alpha);
}

// ---------------- concat [features | g] -> f16 cat[256,1792] ---------------
__global__ void concat_f16(const float* __restrict__ feat,
                           const _Float16* __restrict__ g_h,
                           _Float16* __restrict__ cat_h) {
  int t = blockIdx.x * blockDim.x + threadIdx.x;
  if (t >= N_NODES * (IN_DIM + GH)) return;
  int n = t / (IN_DIM + GH), c = t % (IN_DIM + GH);
  cat_h[t] = (c < IN_DIM) ? (_Float16)feat[n * IN_DIM + c]
                          : g_h[n * GH + (c - IN_DIM)];
}

// ---------------- spatial LN over HC=64 + NHWC->NCHW -----------------------
__global__ void __launch_bounds__(32)
spatial_ln_t(const float* __restrict__ z2, const float* __restrict__ gamma,
             const float* __restrict__ beta, float* __restrict__ zdec) {
  int pos = blockIdx.x;           // n*256 + (y*16+x)
  int n = pos >> 8, p = pos & 255;
  int lane = threadIdx.x;
  float v0 = z2[pos * HC + lane];
  float v1 = z2[pos * HC + 32 + lane];
  float mean = wave_sum32(v0 + v1) * (1.0f / 64.0f);
  float d0 = v0 - mean, d1 = v1 - mean;
  float inv = rsqrtf(wave_sum32(d0 * d0 + d1 * d1) * (1.0f / 64.0f) + 1e-5f);
  int y = p >> 4, x = p & 15;
  int c0 = lane, c1 = lane + 32;
  zdec[((n * HC + c0) * SS + y) * SS + x] = d0 * inv * gamma[c0] + beta[c0];
  zdec[((n * HC + c1) * SS + y) * SS + x] = d1 * inv * gamma[c1] + beta[c1];
}

// ---------------- bilinear up2 sample (half-pixel, edge clamp) -------------
__device__ __forceinline__ float up2_sample(const float* __restrict__ p,
                                            int Hi, int Wi, int y, int x) {
  float fy = 0.5f * (y + 0.5f) - 0.5f;
  float fx = 0.5f * (x + 0.5f) - 0.5f;
  int y0 = (int)floorf(fy), x0 = (int)floorf(fx);
  float wy = fy - y0, wx = fx - x0;
  int y1 = min(Hi - 1, max(0, y0 + 1));
  int x1 = min(Wi - 1, max(0, x0 + 1));
  y0 = min(Hi - 1, max(0, y0));
  x0 = min(Wi - 1, max(0, x0));
  float a = p[y0 * Wi + x0], b = p[y0 * Wi + x1];
  float c = p[y1 * Wi + x0], d = p[y1 * Wi + x1];
  return (1.0f - wy) * ((1.0f - wx) * a + wx * b) +
         wy * ((1.0f - wx) * c + wx * d);
}

// ---------------- conv3x3 (+optional fused up2) + BN(eval) + ReLU ----------
__global__ void conv3x3_bn_relu(const float* __restrict__ in,
                                float* __restrict__ out,
                                const float* __restrict__ w,   // [Co][Ci][3][3]
                                const float* __restrict__ bng,
                                const float* __restrict__ bnb,
                                int N, int Cin, int Cout, int H, int W,
                                int up) {
  int t = blockIdx.x * blockDim.x + threadIdx.x;
  int total = N * Cout * H * W;
  if (t >= total) return;
  int x  = t % W;
  int y  = (t / W) % H;
  int oc = (t / (W * H)) % Cout;
  int n  = t / (W * H * Cout);
  int Hi = up ? H / 2 : H, Wi = up ? W / 2 : W;
  float acc = 0.0f;
  for (int ic = 0; ic < Cin; ++ic) {
    const float* pi = in + ((size_t)(n * Cin + ic)) * Hi * Wi;
    const float* pw = w + ((size_t)(oc * Cin + ic)) * 9;
    if (ic + 1 < Cin) __builtin_prefetch(pi + (size_t)Hi * Wi, 0, 1);
#pragma unroll
    for (int tt = 0; tt < 9; ++tt) {
      int yy = y + tt / 3 - 1, xx = x + tt % 3 - 1;
      float v = 0.0f;
      if (yy >= 0 && yy < H && xx >= 0 && xx < W)  // SAME zero-pad
        v = up ? up2_sample(pi, Hi, Wi, yy, xx) : pi[yy * Wi + xx];
      acc += pw[tt] * v;
    }
  }
  float sc = bng[oc] * rsqrtf(1.0f + 1e-5f);
  out[t] = fmaxf(acc * sc + bnb[oc], 0.0f);
}

// ---------------- fused: up2 + final conv3x3 + BN + ReLU + 1x1 head --------
// in: [N,8,128,128] -> out: [N,3,256,256], never materializing [N,8,256,256].
__global__ void final_head(const float* __restrict__ z,
                           const float* __restrict__ fw,  // [8][8][3][3]
                           const float* __restrict__ fg,
                           const float* __restrict__ fb,
                           const float* __restrict__ hw,  // [3][8]
                           const float* __restrict__ hb,
                           float* __restrict__ out) {
  int t = blockIdx.x * blockDim.x + threadIdx.x;  // n*65536 + y*256 + x
  if (t >= N_NODES * 256 * 256) return;
  int x = t % 256, y = (t / 256) % 256, n = t / 65536;
  float acc[8];
#pragma unroll
  for (int oc = 0; oc < 8; ++oc) acc[oc] = 0.0f;
  for (int ic = 0; ic < 8; ++ic) {
    const float* pi = z + ((size_t)(n * 8 + ic)) * 128 * 128;
    float patch[9];
#pragma unroll
    for (int tt = 0; tt < 9; ++tt) {
      int yy = y + tt / 3 - 1, xx = x + tt % 3 - 1;
      patch[tt] = (yy >= 0 && yy < 256 && xx >= 0 && xx < 256)
                      ? up2_sample(pi, 128, 128, yy, xx)
                      : 0.0f;
    }
#pragma unroll
    for (int oc = 0; oc < 8; ++oc) {
      const float* pw = fw + ((size_t)(oc * 8 + ic)) * 9;
      float a = 0.0f;
#pragma unroll
      for (int tt = 0; tt < 9; ++tt) a += pw[tt] * patch[tt];
      acc[oc] += a;
    }
  }
  float sc = rsqrtf(1.0f + 1e-5f);
  float r[8];
#pragma unroll
  for (int oc = 0; oc < 8; ++oc)
    r[oc] = fmaxf(acc[oc] * fg[oc] * sc + fb[oc], 0.0f);
#pragma unroll
  for (int c = 0; c < 3; ++c) {
    float o = hb[c];
#pragma unroll
    for (int oc = 0; oc < 8; ++oc) o += hw[c * 8 + oc] * r[oc];
    out[((size_t)(n * 3 + c)) * 65536 + y * 256 + x] = o;
  }
}

// ---------------------------------------------------------------------------
// Host orchestration. Input order: flat depth-first in setup_inputs() dict
// insertion order (see Round-0 comment).
// ---------------------------------------------------------------------------
#define IDX_GP   2
#define IDX_GAT  6   // 8 per layer
#define IDX_SP   30
#define IDX_DEC  36  // 6 per block
#define IDX_FIN  54
#define IDX_HEAD 57

static inline int cdiv(int a, int b) { return (a + b - 1) / b; }

extern "C" void kernel_launch(void* const* d_in, const int* in_sizes, int n_in,
                              void* d_out, int out_size, void* d_ws,
                              size_t ws_size, hipStream_t stream) {
  (void)in_sizes; (void)n_in; (void)out_size; (void)ws_size;
  const float* feat = (const float*)d_in[0];
  const int*   eidx = (const int*)d_in[1];
  const float* gpW = (const float*)d_in[IDX_GP + 0];
  const float* gpB = (const float*)d_in[IDX_GP + 1];
  const float* gpG = (const float*)d_in[IDX_GP + 2];
  const float* gpBeta = (const float*)d_in[IDX_GP + 3];
  const float* spW1 = (const float*)d_in[IDX_SP + 0];
  const float* spb1 = (const float*)d_in[IDX_SP + 1];
  const float* spW2 = (const float*)d_in[IDX_SP + 2];
  const float* spb2 = (const float*)d_in[IDX_SP + 3];
  const float* spLg = (const float*)d_in[IDX_SP + 4];
  const float* spLb = (const float*)d_in[IDX_SP + 5];

  // ---- workspace layout ----
  char* ws = (char*)d_ws;
  const size_t PBYTES = (size_t)N_NODES * 8 * 128 * 128 * 4;  // 134,217,728
  float* P0 = (float*)(ws);
  float* P1 = (float*)(ws + PBYTES);
  char*  sm = ws + 2 * PBYTES;
  float*    g      = (float*)(sm);                 // 262144 B
  float*    gtmp   = (float*)(sm + 262144);
  float*    xl     = (float*)(sm + 524288);
  float*    xr     = (float*)(sm + 786432);
  float*    gout   = (float*)(sm + 1048576);
  float*    z1     = (float*)(sm + 1310720);       // 524288 B
  float*    ea     = (float*)(sm + 1835008);
  float*    eex    = (float*)(sm + 1904640);
  unsigned* em     = (unsigned*)(sm + 1974272);
  float*    es     = (float*)(sm + 1978368);
  _Float16* feat_h = (_Float16*)(sm + 1982464);    // 786432 B
  _Float16* gpW_h  = (_Float16*)(sm + 2768896);    // 786432 B
  _Float16* gatW_h = (_Float16*)(sm + 3555328);    // 6*131072 B
  _Float16* W1_h   = (_Float16*)(sm + 4341760);    // 1835008 B
  _Float16* W2_h   = (_Float16*)(sm + 6176768);    // 16777216 B
  _Float16* g_h    = (_Float16*)(sm + 22953984);   // 131072 B
  _Float16* cat_h  = (_Float16*)(sm + 23085056);   // 917504 B
  _Float16* z1_h   = (_Float16*)(sm + 24002560);   // 262144 B

  // ---- one-pass f16 conversion of GEMM operands ----
  cvt_f16<<<cdiv(N_NODES * IN_DIM, 256), 256, 0, stream>>>(feat, feat_h,
                                                           N_NODES * IN_DIM);
  cvt_f16<<<cdiv(IN_DIM * GH, 256), 256, 0, stream>>>(gpW, gpW_h, IN_DIM * GH);
  for (int l = 0; l < 3; ++l) {
    cvt_f16<<<cdiv(GH * GH, 256), 256, 0, stream>>>(
        (const float*)d_in[IDX_GAT + 8 * l + 0], gatW_h + (size_t)l * 2 * GH * GH,
        GH * GH);
    cvt_f16<<<cdiv(GH * GH, 256), 256, 0, stream>>>(
        (const float*)d_in[IDX_GAT + 8 * l + 2],
        gatW_h + (size_t)(l * 2 + 1) * GH * GH, GH * GH);
  }
  cvt_f16<<<cdiv((IN_DIM + GH) * SB, 256), 256, 0, stream>>>(
      spW1, W1_h, (IN_DIM + GH) * SB);
  cvt_f16<<<cdiv(SB * SS * SS * HC, 256), 256, 0, stream>>>(
      spW2, W2_h, SB * SS * SS * HC);

  // 1) graph projection + LN -> g (f32 + f16 shadow)
  wmma_gemm_f16<<<dim3(GH / 32, N_NODES / 32), 32, 0, stream>>>(
      feat_h, IN_DIM, gpW_h, GH, gpB, gtmp, nullptr, GH, IN_DIM, 0);
  ln_row256<<<N_NODES, 32, 0, stream>>>(gtmp, nullptr, nullptr, gpG, gpBeta, g,
                                        g_h);

  // 2) 3x GATv2 (+residual, LN)
  for (int l = 0; l < 3; ++l) {
    const float* bl   = (const float*)d_in[IDX_GAT + 8 * l + 1];
    const float* br   = (const float*)d_in[IDX_GAT + 8 * l + 3];
    const float* att  = (const float*)d_in[IDX_GAT + 8 * l + 4];
    const float* bias = (const float*)d_in[IDX_GAT + 8 * l + 5];
    const float* lng  = (const float*)d_in[IDX_GAT + 8 * l + 6];
    const float* lnb  = (const float*)d_in[IDX_GAT + 8 * l + 7];
    const _Float16* Wl_h = gatW_h + (size_t)l * 2 * GH * GH;
    const _Float16* Wr_h = Wl_h + GH * GH;
    wmma_gemm_f16<<<dim3(8, 8), 32, 0, stream>>>(g_h, GH, Wl_h, GH, bl, xl,
                                                 nullptr, GH, GH, 0);
    wmma_gemm_f16<<<dim3(8, 8), 32, 0, stream>>>(g_h, GH, Wr_h, GH, br, xr,
                                                 nullptr, GH, GH, 0);
    zero_f32<<<cdiv(N_NODES * GH, 256), 256, 0, stream>>>(gout, N_NODES * GH);
    zero_f32<<<4, 256, 0, stream>>>((float*)em, N_NODES * HEADS);
    zero_f32<<<4, 256, 0, stream>>>(es, N_NODES * HEADS);
    const int ET = ETOT * HEADS;
    gat_edge_score<<<cdiv(ET, 256), 256, 0, stream>>>(eidx, xl, xr, att, ea, em);
    gat_edge_exp<<<cdiv(ET, 256), 256, 0, stream>>>(eidx, ea, em, eex, es);
    gat_scatter<<<cdiv(ETOT * GH, 256), 256, 0, stream>>>(eidx, xl, eex, es,
                                                          gout);
    ln_row256<<<N_NODES, 32, 0, stream>>>(gout, g, bias, lng, lnb, g, g_h);
  }

  // 3) spatial MLP (z2 lives in P0)
  concat_f16<<<cdiv(N_NODES * (IN_DIM + GH), 256), 256, 0, stream>>>(feat, g_h,
                                                                     cat_h);
  wmma_gemm_f16<<<dim3(SB / 32, N_NODES / 32), 32, 0, stream>>>(
      cat_h, IN_DIM + GH, W1_h, SB, spb1, z1, z1_h, SB, IN_DIM + GH, 1);
  wmma_gemm_f16<<<dim3((SS * SS * HC) / 32, N_NODES / 32), 32, 0, stream>>>(
      z1_h, SB, W2_h, SS * SS * HC, spb2, P0, nullptr, SS * SS * HC, SB, 0);
  spatial_ln_t<<<N_NODES * SS * SS, 32, 0, stream>>>(P0, spLg, spLb, P1);

  // 4) decoder: (up2+conv+BN+ReLU) + (conv+BN+ReLU), ping-pong P1<->P0
  struct { int ci, co, h; } cfg[3] = {{64, 32, 32}, {32, 16, 64}, {16, 8, 128}};
  float* bufs[2] = {P1, P0};
  int cur = 0;  // zdec currently in P1
  for (int b = 0; b < 3; ++b) {
    const float* w1 = (const float*)d_in[IDX_DEC + 6 * b + 0];
    const float* g1 = (const float*)d_in[IDX_DEC + 6 * b + 1];
    const float* b1 = (const float*)d_in[IDX_DEC + 6 * b + 2];
    const float* w2 = (const float*)d_in[IDX_DEC + 6 * b + 3];
    const float* g2 = (const float*)d_in[IDX_DEC + 6 * b + 4];
    const float* b2 = (const float*)d_in[IDX_DEC + 6 * b + 5];
    int H = cfg[b].h;
    int tot = N_NODES * cfg[b].co * H * H;
    conv3x3_bn_relu<<<cdiv(tot, 256), 256, 0, stream>>>(
        bufs[cur], bufs[cur ^ 1], w1, g1, b1, N_NODES, cfg[b].ci, cfg[b].co, H,
        H, 1);
    cur ^= 1;
    conv3x3_bn_relu<<<cdiv(tot, 256), 256, 0, stream>>>(
        bufs[cur], bufs[cur ^ 1], w2, g2, b2, N_NODES, cfg[b].co, cfg[b].co, H,
        H, 0);
    cur ^= 1;
  }

  // 5) fused up2 + final conv + BN + ReLU + 1x1 head -> d_out
  const float* fw = (const float*)d_in[IDX_FIN + 0];
  const float* fg = (const float*)d_in[IDX_FIN + 1];
  const float* fb = (const float*)d_in[IDX_FIN + 2];
  const float* hw = (const float*)d_in[IDX_HEAD + 0];
  const float* hb = (const float*)d_in[IDX_HEAD + 1];
  final_head<<<cdiv(N_NODES * 256 * 256, 256), 256, 0, stream>>>(
      bufs[cur], fw, fg, fb, hw, hb, (float*)d_out);
}
